// SwinTransformerBlock_15693810500360
// MI455X (gfx1250) — matrix operand order repaired and checked by reading, not compiled, not run on hardware
//
#include <hip/hip_runtime.h>
#include <math.h>

// ---------------------------------------------------------------------------
// Types
// ---------------------------------------------------------------------------
typedef __bf16 bf16_t;
typedef __bf16 v16bf __attribute__((ext_vector_type(16)));
typedef __bf16 v8bf  __attribute__((ext_vector_type(8)));
typedef __bf16 v4bf  __attribute__((ext_vector_type(4)));
typedef float  v8f   __attribute__((ext_vector_type(8)));
typedef unsigned int v4u __attribute__((ext_vector_type(4)));
typedef int v8i __attribute__((ext_vector_type(8)));
typedef int v4i __attribute__((ext_vector_type(4)));

#define D_DIM   2048
#define FF_DIM  8192
#define NTOK    1024   // B * NW
#define HDIM    256    // D / NH

// ---------------------------------------------------------------------------
// Tensor Data Mover: build a 2D D# (bf16 elements) and issue
// tensor_load_to_lds (6-arg clang-23 form). Descriptor layout per CDNA5 ISA
// 08_async_tensor §8.
//   group0: [1:0]=count=1, [63:32]=lds_addr, [120:64]=global_addr, [127:126]=2
//   group1: data_size=1(2B), pad fields, tensor_dim0/1, tile_dim0/1,
//           tensor_dim0_stride
// LDS row padding is done by the TDM itself: pad_interval code c = 2^(c+1)
// DWORDs between pads, pad_amount code a = (a+1) DWORDs of pad.
// ---------------------------------------------------------------------------
__device__ __forceinline__ void tdm_load_2d(unsigned lds_addr, const void* gaddr,
                                            unsigned tile_d0, unsigned tile_d1,
                                            unsigned tensor_d0, unsigned tensor_d1,
                                            unsigned stride_elems,
                                            unsigned pad_interval_code,
                                            unsigned pad_amount_code) {
    unsigned long long ga = (unsigned long long)(size_t)gaddr;
    v4u g0;
    g0[0] = 1u;                                        // count=1, user descriptor
    g0[1] = lds_addr;                                  // LDS byte address
    g0[2] = (unsigned)(ga & 0xFFFFFFFFu);              // global_addr[31:0]
    g0[3] = (unsigned)((ga >> 32) & 0x01FFFFFFu) | (2u << 30);  // addr[56:32], type=2
    v8i g1;
    g1[0] = (int)((1u << 16) |                         // data_size = 2 bytes
                  (1u << 20) |                         // pad_enable
                  (pad_interval_code << 22) |
                  (pad_amount_code << 25));
    g1[1] = (int)((tensor_d0 & 0xFFFFu) << 16);        // tensor_dim0[15:0]
    g1[2] = (int)((tensor_d0 >> 16) | ((tensor_d1 & 0xFFFFu) << 16));
    g1[3] = (int)((tensor_d1 >> 16) | (tile_d0 << 16));// tile_dim0
    g1[4] = (int)(tile_d1 & 0xFFFFu);                  // tile_dim1 (tile_dim2=0)
    g1[5] = (int)stride_elems;                         // tensor_dim0_stride[31:0]
    g1[6] = 0;                                         // stride hi, dim1_stride lo
    g1[7] = 0;
    v4i g2 = {0, 0, 0, 0};
    v4i g3 = {0, 0, 0, 0};
    v8i g4 = {0, 0, 0, 0, 0, 0, 0, 0};
    __builtin_amdgcn_tensor_load_to_lds(g0, g1, g2, g3, g4, 0);
}

// ---------------------------------------------------------------------------
// f32 -> bf16 weight conversion (one pass; bf16 set then lives in 192MB L2)
// ---------------------------------------------------------------------------
__global__ void cvt_f32_to_bf16(const float* __restrict__ src,
                                bf16_t* __restrict__ dst, int n) {
    int i = (blockIdx.x * blockDim.x + threadIdx.x) * 4;
    if (i + 4 <= n) {
        float4 v = *(const float4*)(src + i);
        v4bf o;
        o[0] = (bf16_t)v.x; o[1] = (bf16_t)v.y;
        o[2] = (bf16_t)v.z; o[3] = (bf16_t)v.w;
        *(v4bf*)(dst + i) = o;
    } else {
        for (int k = i; k < n; ++k) dst[k] = (bf16_t)src[k];
    }
}

// ---------------------------------------------------------------------------
// LayerNorm. MODE 0: window-partition embeddings -> windows(f32) + ln(bf16)
//           MODE 1: plain row LN of f32 src -> ln(bf16)
// ---------------------------------------------------------------------------
template <int MODE>
__global__ __launch_bounds__(256)
void layernorm_k(const float* __restrict__ src,
                 const float* __restrict__ gamma,
                 const float* __restrict__ beta,
                 float* __restrict__ win_out,
                 bf16_t* __restrict__ ln_out) {
    const int row = blockIdx.x, tid = threadIdx.x;
    float vals[8];
    if (MODE == 0) {
        const int b = row >> 6, n = row & 63;
        const int vw = n >> 3, hw = n & 7;
        for (int j = 0; j < 8; ++j) {
            const int d = tid + j * 256;
            const int p = d >> 5, e = d & 31;
            const int vp = p >> 3, hp = p & 7;
            vals[j] = src[((size_t)(b * 64 + vp * 8 + vw) * 64 + (hp * 8 + hw)) * 32 + e];
        }
    } else {
        for (int j = 0; j < 8; ++j)
            vals[j] = src[(size_t)row * D_DIM + tid + j * 256];
    }
    float s = 0.f, q = 0.f;
    for (int j = 0; j < 8; ++j) { s += vals[j]; q += vals[j] * vals[j]; }

    __shared__ float rs[256], rq[256];
    rs[tid] = s; rq[tid] = q;
    __syncthreads();
    for (int off = 128; off > 0; off >>= 1) {
        if (tid < off) { rs[tid] += rs[tid + off]; rq[tid] += rq[tid + off]; }
        __syncthreads();
    }
    const float mean = rs[0] * (1.0f / 2048.0f);
    const float var  = rq[0] * (1.0f / 2048.0f) - mean * mean;
    const float inv  = rsqrtf(var + 1e-5f);

    for (int j = 0; j < 8; ++j) {
        const int d = tid + j * 256;
        const float xn = (vals[j] - mean) * inv * gamma[d] + beta[d];
        ln_out[(size_t)row * D_DIM + d] = (bf16_t)xn;
        if (MODE == 0) win_out[(size_t)row * D_DIM + d] = vals[j];
    }
}

// ---------------------------------------------------------------------------
// bf16 WMMA GEMM with TDM-staged, double-buffered LDS tiles.
// Block tile 128x128, k-step 32. 8 wave32s; wave (wm,wn) in 2x4 grid owns a
// 64x32 patch = 4x2 v_wmma_f32_16x16x32_bf16 accumulators.
// Wave 0 issues tensor_load_to_lds for the NEXT k-tile while all waves
// compute on the current one; per-iteration sync = s_wait_tensorcnt + barrier.
// TDM pads LDS rows: A 32h rows +8h (16 DW -> code 3, 4 DW pad -> code 3),
//                    B 128h rows +8h (64 DW -> code 5, 4 DW pad -> code 3).
// ---------------------------------------------------------------------------
enum { EPI_BF16 = 0, EPI_GELU = 1, EPI_RES = 2 };

template <int EPI>
__global__ __launch_bounds__(256)
void gemm_bf16_k(const bf16_t* __restrict__ A, const bf16_t* __restrict__ Bm,
                 const float* __restrict__ bias,
                 bf16_t* __restrict__ outB, const float* __restrict__ res,
                 float* __restrict__ outF, int M, int N, int K) {
    __shared__ bf16_t As[2][128][40];   // 128x32 tile, stride 40 halfs (80B = 5*16)
    __shared__ bf16_t Bs[2][32][136];   // 32x128 tile, stride 136 halfs (272B = 17*16)

    const int tid = threadIdx.x, lane = tid & 31, wid = tid >> 5;
    const int wm = wid >> 2, wn = wid & 3;
    const int m0 = blockIdx.y * 128, n0 = blockIdx.x * 128;

    v8f zero = {0.f, 0.f, 0.f, 0.f, 0.f, 0.f, 0.f, 0.f};
    v8f acc[4][2];
    for (int i = 0; i < 4; ++i)
        for (int j = 0; j < 2; ++j) acc[i][j] = zero;

    const unsigned ldsA[2] = {(unsigned)(size_t)&As[0][0][0],
                              (unsigned)(size_t)&As[1][0][0]};
    const unsigned ldsB[2] = {(unsigned)(size_t)&Bs[0][0][0],
                              (unsigned)(size_t)&Bs[1][0][0]};
    const bf16_t* Abase = A + (size_t)m0 * K;        // tile kk at +kk elems
    const bf16_t* Bbase = Bm + n0;                   // tile kk at +kk*N elems

    // prologue: TDM loads for k-tile 0 into buffer 0
    if (tid < 32) {
        tdm_load_2d(ldsA[0], Abase, 32, 128, (unsigned)K, (unsigned)M,
                    (unsigned)K, 3u, 3u);
        tdm_load_2d(ldsB[0], Bbase, 128, 32, (unsigned)N, (unsigned)K,
                    (unsigned)N, 5u, 3u);
    }

    int cur = 0;
    for (int kk = 0; kk < K; kk += 32) {
        if (tid < 32) __builtin_amdgcn_s_wait_tensorcnt(0);
        __syncthreads();   // buf[cur] ready; everyone done reading buf[cur^1]
        if (tid < 32 && kk + 32 < K) {
            tdm_load_2d(ldsA[cur ^ 1], Abase + (size_t)(kk + 32), 32, 128,
                        (unsigned)K, (unsigned)M, (unsigned)K, 3u, 3u);
            tdm_load_2d(ldsB[cur ^ 1], Bbase + (size_t)(kk + 32) * N, 128, 32,
                        (unsigned)N, (unsigned)K, (unsigned)N, 5u, 3u);
        }

        // A fragment: lane l -> row M = l&15, K-halves {b..b+7, b+16..b+23}
        v16bf afrag[4], bfrag[2];
        const int mloc = wm * 64 + (lane & 15);
        const int kb = (lane >> 4) * 8;
        for (int i = 0; i < 4; ++i) {
            const bf16_t* p = &As[cur][mloc + i * 16][kb];
            v8bf lo = *(const v8bf*)p;
            v8bf hi = *(const v8bf*)(p + 16);
            afrag[i] = __builtin_shufflevector(lo, hi,
                0, 1, 2, 3, 4, 5, 6, 7, 8, 9, 10, 11, 12, 13, 14, 15);
        }
        // B fragment: lane l -> col N = l&15, K-halves {b..b+7, b+16..b+23}
        for (int j = 0; j < 2; ++j) {
            const bf16_t* p = &Bs[cur][kb][wn * 32 + j * 16 + (lane & 15)];
            for (int s = 0; s < 16; ++s)
                bfrag[j][s] = p[(s < 8 ? s : s + 8) * 136];
        }
        for (int i = 0; i < 4; ++i)
            for (int j = 0; j < 2; ++j)
                acc[i][j] = __builtin_amdgcn_wmma_f32_16x16x32_bf16(
                    false, afrag[i], false, bfrag[j], (short)0, acc[i][j],
                    false, false);
        cur ^= 1;
    }

    // Epilogue. C layout: VGPR r -> M = r (+8 for lanes 16-31), N = lane&15.
    const int rbase = (lane >> 4) * 8;
    const int nl = lane & 15;
    for (int i = 0; i < 4; ++i) {
        for (int j = 0; j < 2; ++j) {
            const int gm = m0 + wm * 64 + i * 16;
            const int gn = n0 + wn * 32 + j * 16 + nl;
            const float bv = bias[gn];
            for (int r = 0; r < 8; ++r) {
                const int row = gm + r + rbase;
                float v = acc[i][j][r] + bv;
                if (EPI == EPI_GELU) {
                    v = 0.5f * v * (1.0f + erff(v * 0.70710678118654752f));
                    outB[(size_t)row * N + gn] = (bf16_t)v;
                } else if (EPI == EPI_BF16) {
                    outB[(size_t)row * N + gn] = (bf16_t)v;
                } else {
                    outF[(size_t)row * N + gn] = v + res[(size_t)row * N + gn];
                }
            }
        }
    }
}

// ---------------------------------------------------------------------------
// Attention: one block per (batch, head). K/V tiles staged via TDM
// (64 rows x 256 halfs, stride 2048; LDS pad 4 DW per 128 DW -> 264-half rows).
// 64x64 logits, diag mask, softmax, P@V, fused residual. ~0.5% of FLOPs.
// ---------------------------------------------------------------------------
__global__ __launch_bounds__(256)
void attention_k(const bf16_t* __restrict__ Q, const bf16_t* __restrict__ Kb,
                 const bf16_t* __restrict__ Vb, const float* __restrict__ pos_bias,
                 const float* __restrict__ windows, float* __restrict__ attn_out) {
    __shared__ bf16_t KV[64][264];   // K tile, later reused for V
    __shared__ float  S[64][65];     // logits / probabilities
    const int tid = threadIdx.x;
    const int b = blockIdx.x >> 3, h = blockIdx.x & 7;
    const size_t base = (size_t)b * 64 * D_DIM + (size_t)h * HDIM;
    const unsigned ldsKV = (unsigned)(size_t)&KV[0][0];

    // stage K[64][256] via TDM (wave 0 issues, TENSORcnt gates the barrier)
    if (tid < 32) {
        tdm_load_2d(ldsKV, Kb + base, 256, 64, 2048u, 1024u, 2048u, 6u, 3u);
        __builtin_amdgcn_s_wait_tensorcnt(0);
    }
    __syncthreads();

    // logits = Q K^T / sqrt(2048) + pos_bias; diagonal masked
    for (int s = tid; s < 64 * 64; s += 256) {
        const int n = s >> 6, m = s & 63;
        float acc;
        if (m == n) {
            acc = -1e30f;
        } else {
            acc = 0.f;
            const bf16_t* qp = Q + base + (size_t)n * D_DIM;
            #pragma unroll 8
            for (int d = 0; d < 256; ++d)
                acc += (float)qp[d] * (float)KV[m][d];
            acc = acc * 0.022097086912079608f + pos_bias[n * 64 + m];
        }
        S[n][m] = acc;
    }
    __syncthreads();

    // stage V over K (TDM), softmax rows in parallel on other lanes
    if (tid < 32) {
        tdm_load_2d(ldsKV, Vb + base, 256, 64, 2048u, 1024u, 2048u, 6u, 3u);
        __builtin_amdgcn_s_wait_tensorcnt(0);
    }
    if (tid < 64) {
        const int n = tid;
        float mx = -1e30f;
        for (int m = 0; m < 64; ++m) mx = fmaxf(mx, S[n][m]);
        float sum = 0.f;
        for (int m = 0; m < 64; ++m) {
            const float e = __expf(S[n][m] - mx);
            S[n][m] = e; sum += e;
        }
        const float inv = 1.f / sum;
        for (int m = 0; m < 64; ++m) S[n][m] *= inv;
    }
    __syncthreads();

    // O = P @ V, fused residual: attn_out = windows + O
    for (int f = tid; f < 64 * 256; f += 256) {
        const int n = f >> 8, d = f & 255;
        float acc = 0.f;
        #pragma unroll 8
        for (int m = 0; m < 64; ++m) acc += S[n][m] * (float)KV[m][d];
        const size_t idx = base + (size_t)n * D_DIM + d;
        attn_out[idx] = windows[idx] + acc;
    }
}

// ---------------------------------------------------------------------------
// Launch
// ---------------------------------------------------------------------------
extern "C" void kernel_launch(void* const* d_in, const int* in_sizes, int n_in,
                              void* d_out, int out_size, void* d_ws, size_t ws_size,
                              hipStream_t stream) {
    (void)in_sizes; (void)n_in; (void)out_size; (void)ws_size;
    const float* emb  = (const float*)d_in[0];
    const float* ln1g = (const float*)d_in[1];
    const float* ln1b = (const float*)d_in[2];
    const float* wq   = (const float*)d_in[3];
    const float* bq   = (const float*)d_in[4];
    const float* wk   = (const float*)d_in[5];
    const float* bk   = (const float*)d_in[6];
    const float* wv   = (const float*)d_in[7];
    const float* bv   = (const float*)d_in[8];
    const float* posb = (const float*)d_in[9];
    const float* ln2g = (const float*)d_in[10];
    const float* ln2b = (const float*)d_in[11];
    const float* w1   = (const float*)d_in[12];
    const float* b1   = (const float*)d_in[13];
    const float* w2   = (const float*)d_in[14];
    const float* b2   = (const float*)d_in[15];

    char* ws = (char*)d_ws;
    const size_t MB = 1u << 20;
    float*  windows  = (float*)(ws + 0);           // 8 MB
    float*  attn_out = (float*)(ws + 8 * MB);      // 8 MB
    bf16_t* xln  = (bf16_t*)(ws + 16 * MB);        // 4 MB
    bf16_t* yln  = (bf16_t*)(ws + 20 * MB);        // 4 MB
    bf16_t* Qb   = (bf16_t*)(ws + 24 * MB);        // 4 MB
    bf16_t* Kbuf = (bf16_t*)(ws + 28 * MB);        // 4 MB
    bf16_t* Vbuf = (bf16_t*)(ws + 32 * MB);        // 4 MB
    bf16_t* hbuf = (bf16_t*)(ws + 36 * MB);        // 16 MB
    bf16_t* Wqb  = (bf16_t*)(ws + 52 * MB);        // 8 MB
    bf16_t* Wkb  = (bf16_t*)(ws + 60 * MB);        // 8 MB
    bf16_t* Wvb  = (bf16_t*)(ws + 68 * MB);        // 8 MB
    bf16_t* W1b  = (bf16_t*)(ws + 76 * MB);        // 32 MB
    bf16_t* W2b  = (bf16_t*)(ws + 108 * MB);       // 32 MB -> 140 MB total

    // 1) weights to bf16 (entire 88MB bf16 set fits in 192MB L2)
    cvt_f32_to_bf16<<<4096, 256, 0, stream>>>(wq, Wqb, D_DIM * D_DIM);
    cvt_f32_to_bf16<<<4096, 256, 0, stream>>>(wk, Wkb, D_DIM * D_DIM);
    cvt_f32_to_bf16<<<4096, 256, 0, stream>>>(wv, Wvb, D_DIM * D_DIM);
    cvt_f32_to_bf16<<<16384, 256, 0, stream>>>(w1, W1b, D_DIM * FF_DIM);
    cvt_f32_to_bf16<<<16384, 256, 0, stream>>>(w2, W2b, FF_DIM * D_DIM);

    // 2) window partition + LN1
    layernorm_k<0><<<NTOK, 256, 0, stream>>>(emb, ln1g, ln1b, windows, xln);

    // 3) Q/K/V projections (WMMA bf16, TDM staging)
    gemm_bf16_k<EPI_BF16><<<dim3(16, 8), 256, 0, stream>>>(
        xln, Wqb, bq, Qb, nullptr, nullptr, NTOK, D_DIM, D_DIM);
    gemm_bf16_k<EPI_BF16><<<dim3(16, 8), 256, 0, stream>>>(
        xln, Wkb, bk, Kbuf, nullptr, nullptr, NTOK, D_DIM, D_DIM);
    gemm_bf16_k<EPI_BF16><<<dim3(16, 8), 256, 0, stream>>>(
        xln, Wvb, bv, Vbuf, nullptr, nullptr, NTOK, D_DIM, D_DIM);

    // 4) attention + residual
    attention_k<<<128, 256, 0, stream>>>(Qb, Kbuf, Vbuf, posb, windows, attn_out);

    // 5) LN2
    layernorm_k<1><<<NTOK, 256, 0, stream>>>(attn_out, ln2g, ln2b, nullptr, yln);

    // 6) FFN1 with fused exact GELU
    gemm_bf16_k<EPI_GELU><<<dim3(64, 8), 256, 0, stream>>>(
        yln, W1b, b1, hbuf, nullptr, nullptr, NTOK, FF_DIM, D_DIM);

    // 7) FFN2 with fused bias + residual -> final f32 output
    gemm_bf16_k<EPI_RES><<<dim3(16, 8), 256, 0, stream>>>(
        hbuf, W2b, b2, nullptr, attn_out, (float*)d_out, NTOK, D_DIM, FF_DIM);
}
